// MCorrLCorr_77266461655742
// MI455X (gfx1250) — matrix operand order repaired and checked by reading, not compiled, not run on hardware
//
#include <hip/hip_runtime.h>

typedef __attribute__((ext_vector_type(16))) __bf16 v16bf;
typedef __attribute__((ext_vector_type(8)))  float  v8f;
typedef __attribute__((ext_vector_type(2)))  unsigned uint2v;

#define BATCH 16
#define CIN   32
#define NGX   128
#define NGY   384
#define COUT  64
#define NFX   4
#define NFY   8
#define NHX   32
#define NHY   190
#define KROW  (NFX*NFY)                 // 32 k-values per channel
#define TILE_HY 48
#define NT    3                         // hy subtiles per WG (and per wave)
#define N_HY_TILES 4                    // ceil(190/48)
#define WIN   (2*TILE_HY + NFY - 2)     // 102 gy values per (c, gx-row)
#define SE    104                       // run stride (bf16): lane-half offset = 208 dw == 16 mod 64 banks
#define CSTR  (NFX*2*SE)                // bf16 elems per channel in LDS
#define CBYTES (2*CSTR)                 // 1664 B per channel
#define THREADS 128                     // 4 waves: one per o-tile; each wave does all 3 hy-subtiles
#define FRAG_V16 (CIN*4*32)
#define FRAG_ELEMS (FRAG_V16*16)        // 65536 bf16 per plane

// dual-dword LDS load with guaranteed immediate offsets (dword units)
#define LD2(dst, o0, o1) \
    asm volatile("ds_load_2addr_b32 %0, %1 offset0:%c2 offset1:%c3" \
                 : "=v"(dst) : "v"(ba), "i"(o0), "i"(o1))
// counted DS wait, data-ordered via pass-through operands
#define DSWAIT(a, n) \
    asm volatile("s_wait_dscnt %c8" \
                 : "+v"(a[0]), "+v"(a[1]), "+v"(a[2]), "+v"(a[3]), \
                   "+v"(a[4]), "+v"(a[5]), "+v"(a[6]), "+v"(a[7]) : "i"(n))

// ---------------------------------------------------------------------------
// Prep: split weight f32 -> bf16 hi/lo in WMMA A-fragment order:
//   plane[((c*4 + mt)*32 + lane)*16 + e],  k(e) = 16*(lane>=16) + e
// ---------------------------------------------------------------------------
__global__ __launch_bounds__(256)
void prep_weight_frag(const float* __restrict__ gw,
                      __bf16* __restrict__ wHi, __bf16* __restrict__ wLo)
{
    const int idx  = blockIdx.x * 256 + threadIdx.x;   // 0 .. 65535
    const int e    = idx & 15;
    const int lane = (idx >> 4) & 31;
    const int mt   = (idx >> 9) & 3;
    const int c    = idx >> 11;
    const int kA   = (lane >> 4) * 16;
    const int k    = kA + e;
    const int o    = mt * 16 + (lane & 15);
    const float x  = gw[o * (CIN * KROW) + c * KROW + k];
    const __bf16 h = (__bf16)x;
    wHi[idx] = h;
    wLo[idx] = (__bf16)(x - (float)h);
}

// ---------------------------------------------------------------------------
// Main: implicit GEMM, M=64 (c_out), N=(b,hx,hy), K=1024 (c,fx,fy),
// bf16x3 split-precision WMMA. Weight read exactly once per WG; B loads via
// inline ds asm with immediate offsets + counted dscnt waits.
// ---------------------------------------------------------------------------
union V16 { unsigned u[8]; v16bf v; };

__global__ __launch_bounds__(THREADS)
void mcorrlcorr_bf16x3_wmma(const float* __restrict__ gin,
                            const __bf16* __restrict__ wHi,
                            const __bf16* __restrict__ wLo,
                            const float* __restrict__ gbias,
                            float* __restrict__ gout)
{
    // interleaved layout: s[(((c*NFX)+fx)*2 + plane)*SE + y], plane 0=hi 1=lo
    __shared__ __bf16 sB[CIN * CSTR];   // 53,248 B

    const int hy0 = blockIdx.x * TILE_HY;
    const int hx  = blockIdx.y;
    const int b   = blockIdx.z;
    const int gy0 = 2 * hy0 - 2;        // gy = 2*hy + fy - 2
    const int tid = threadIdx.x;

    // ---- stage + split input window (each element converted exactly once) ----
    for (int idx = tid; idx < CIN * NFX * WIN; idx += THREADS) {
        const int c   = idx / (NFX * WIN);
        const int rem = idx % (NFX * WIN);
        const int r   = rem / WIN;
        const int y   = rem % WIN;
        const int gx  = (hx + 1) * (r + 1) - 1;      // always < 128 for hx < 32
        const int gy  = gy0 + y;
        float x = 0.0f;
        if (gy >= 0 && gy < NGY)
            x = gin[(((size_t)b * CIN + c) * NGX + gx) * NGY + gy];
        const __bf16 h = (__bf16)x;
        __bf16* p = sB + (c * NFX + r) * (2 * SE) + y;
        p[0]  = h;                                   // hi run
        p[SE] = (__bf16)(x - (float)h);              // lo run
    }
    __syncthreads();

    const int mt   = tid >> 5;         // o tile: M = mt*16 .. +15 (one per wave)
    const int lane = tid & 31;
    const int lr   = lane & 15;
    const int hh   = lane >> 4;        // lane half

    // A fragments: fragment-ordered planes, stride per channel = 128 v16bf
    const v16bf* fHi = ((const v16bf*)wHi) + (size_t)mt * 32 + lane;
    const v16bf* fLo = ((const v16bf*)wLo) + (size_t)mt * 32 + lane;

    // B base: LDS byte address (wave-relative, per ISA flat-LDS aperture rule)
    unsigned ba = (unsigned)(uintptr_t)(sB + (2 * hh) * (2 * SE) + 2 * lr);

    v8f accHH[NT], accLH[NT], accHL[NT];
    #pragma unroll
    for (int t = 0; t < NT; ++t) { accHH[t] = (v8f){}; accLH[t] = (v8f){}; accHL[t] = (v8f){}; }

    #pragma unroll 1
    for (int c = 0; c < CIN; ++c) {
        const v16bf aHi = fHi[c * 128];     // 2x global_load_b128, imm offsets, L2-resident
        const v16bf aLo = fLo[c * 128];

        // issue all 24 B loads (in-order dscnt); offsets <= 191 dwords
        uint2v r[NT][8];
        #pragma unroll
        for (int t = 0; t < NT; ++t) {
            LD2(r[t][0], t*16 + 0,          t*16 + 1);           // BH fx run 0
            LD2(r[t][1], t*16 + 2,          t*16 + 3);
            LD2(r[t][2], t*16 + SE + 0,     t*16 + SE + 1);      // BH fx run 1
            LD2(r[t][3], t*16 + SE + 2,     t*16 + SE + 3);
            LD2(r[t][4], t*16 + SE/2 + 0,   t*16 + SE/2 + 1);    // BL fx run 0
            LD2(r[t][5], t*16 + SE/2 + 2,   t*16 + SE/2 + 3);
            LD2(r[t][6], t*16 + 3*SE/2 + 0, t*16 + 3*SE/2 + 1);  // BL fx run 1
            LD2(r[t][7], t*16 + 3*SE/2 + 2, t*16 + 3*SE/2 + 3);
        }
        ba += CBYTES;                       // one induction add per channel

        #pragma unroll
        for (int t = 0; t < NT; ++t) {
            DSWAIT(r[t], (NT - 1 - t) * 8); // counted wait: later subtiles stay in flight
            V16 BH, BL;
            BH.u[0]=r[t][0].x; BH.u[1]=r[t][0].y; BH.u[2]=r[t][1].x; BH.u[3]=r[t][1].y;
            BH.u[4]=r[t][2].x; BH.u[5]=r[t][2].y; BH.u[6]=r[t][3].x; BH.u[7]=r[t][3].y;
            BL.u[0]=r[t][4].x; BL.u[1]=r[t][4].y; BL.u[2]=r[t][5].x; BL.u[3]=r[t][5].y;
            BL.u[4]=r[t][6].x; BL.u[5]=r[t][6].y; BL.u[6]=r[t][7].x; BL.u[7]=r[t][7].y;
            accHH[t] = __builtin_amdgcn_wmma_f32_16x16x32_bf16(false, aHi, false, BH.v, (short)0, accHH[t], false, false);
            accLH[t] = __builtin_amdgcn_wmma_f32_16x16x32_bf16(false, aLo, false, BH.v, (short)0, accLH[t], false, false);
            accHL[t] = __builtin_amdgcn_wmma_f32_16x16x32_bf16(false, aHi, false, BL.v, (short)0, accHL[t], false, false);
        }
    }

    // ---- store D + bias (C/D layout: VGPR v -> M = v + 8*hh, N = lane%16) ----
    #pragma unroll
    for (int t = 0; t < NT; ++t) {
        const int hy = hy0 + t * 16 + lr;
        if (hy < NHY) {
            #pragma unroll
            for (int v = 0; v < 8; ++v) {
                const int oo = mt * 16 + hh * 8 + v;
                const float res = accHH[t][v] + accLH[t][v] + accHL[t][v] + gbias[oo];
                gout[(((size_t)b * COUT + oo) * NHX + hx) * NHY + hy] = res;
            }
        }
    }
}

extern "C" void kernel_launch(void* const* d_in, const int* in_sizes, int n_in,
                              void* d_out, int out_size, void* d_ws, size_t ws_size,
                              hipStream_t stream) {
    (void)in_sizes; (void)n_in; (void)ws_size; (void)out_size;
    const float* gin = (const float*)d_in[0];   // [16,32,128,384] f32
    const float* gw  = (const float*)d_in[1];   // [64,32,4,8]     f32
    const float* gb  = (const float*)d_in[2];   // [64]            f32
    float* gout = (float*)d_out;                // [16,64,32,190]  f32

    __bf16* wHi = (__bf16*)d_ws;                // 128 KB
    __bf16* wLo = wHi + FRAG_ELEMS;             // 128 KB (256 KB scratch total)

    prep_weight_frag<<<dim3(FRAG_ELEMS / 256), 256, 0, stream>>>(gw, wHi, wLo);

    dim3 grid(N_HY_TILES, NHX, BATCH);          // 4 x 32 x 16 = 2048 WGs
    mcorrlcorr_bf16x3_wmma<<<grid, THREADS, 0, stream>>>(gin, wHi, wLo, gb, gout);
}
